// TemporalDiscriminator_40157944218357
// MI455X (gfx1250) — compile-verified, matrix operand order checked by reading, not run on hardware
//
#include <hip/hip_runtime.h>
#include <hip/hip_bf16.h>

// ---------------- problem constants ----------------
#define BATCH 64
#define SEQ   512
#define DIN   2818
#define DTXT  512
#define HDIM  256
#define NG    1024   // 4*H
#define KREC  512    // H (x part) + H (h part)
#define BS    (BATCH*SEQ)   // 32768
#define NCHUNK ((DIN + 31) / 32)   // 89 (88 full + 1 two-column tail)

// ---------------- WMMA fragment types ----------------
typedef __attribute__((ext_vector_type(16))) __bf16 v16bf;
typedef __attribute__((ext_vector_type(8)))  float  v8f;
typedef __attribute__((ext_vector_type(4)))  unsigned int v4u;
typedef __attribute__((ext_vector_type(2)))  __bf16 bf16x2;
typedef __attribute__((ext_vector_type(2)))  float  f32x2;

union FragAB { v16bf v; v4u q[2]; unsigned short u[16]; };
union FragC  { v8f  v; float f[8]; };

__device__ inline unsigned short f2bf(float f) {
  union { __bf16 h; unsigned short u; } cv;
  cv.h = (__bf16)f;          // hardware RNE f32->bf16 convert
  return cv.u;
}
// two floats -> packed bf16x2 in one v_cvt_pk_bf16_f32
__device__ inline unsigned int pack_bf2(float lo, float hi) {
  f32x2 f; f.x = lo; f.y = hi;
  union { bf16x2 h; unsigned int u; } cv;
  cv.h = __builtin_convertvector(f, bf16x2);
  return cv.u;
}
__device__ inline float bf2f(unsigned short h) {
  return __uint_as_float(((unsigned int)h) << 16);
}
__device__ inline float sigf(float x) { return 1.f / (1.f + __expf(-x)); }

// =====================================================================
// Kernel 1: pack LSTM weights to bf16  Wcomb[dir][n][k] = k<256?Wih:Whh
//           and combined biases bcomb[dir][n] = bih[n]+bhh[n]
// =====================================================================
__global__ __launch_bounds__(256)
void k_prep(const float* __restrict__ Wih_f, const float* __restrict__ Whh_f,
            const float* __restrict__ bih_f, const float* __restrict__ bhh_f,
            const float* __restrict__ Wih_b, const float* __restrict__ Whh_b,
            const float* __restrict__ bih_b, const float* __restrict__ bhh_b,
            unsigned short* __restrict__ Wcomb, float* __restrict__ bcomb) {
  // process pairs: k even, convert two at a time
  const int totalp = (2 * NG * KREC) / 2;        // 524,288 pairs
  for (int pidx = blockIdx.x * blockDim.x + threadIdx.x; pidx < totalp;
       pidx += gridDim.x * blockDim.x) {
    int idx = pidx * 2;
    int dir = idx >> 19;                         // NG*KREC = 2^19
    int rem = idx & ((NG * KREC) - 1);
    int n = rem >> 9;                            // /KREC
    int k = rem & (KREC - 1);                    // even
    const float* Wih = dir ? Wih_b : Wih_f;
    const float* Whh = dir ? Whh_b : Whh_f;
    // pair never straddles the 256 boundary (k even, pair within same half)
    const float* src = (k < HDIM) ? (Wih + n * HDIM + k)
                                  : (Whh + n * HDIM + (k - HDIM));
    *(unsigned int*)&Wcomb[idx] = pack_bf2(src[0], src[1]);
  }
  int t = blockIdx.x * blockDim.x + threadIdx.x;
  if (t < 2 * NG) {
    int dir = t >> 10, n = t & (NG - 1);
    const float* b1 = dir ? bih_b : bih_f;
    const float* b2 = dir ? bhh_b : bhh_f;
    bcomb[t] = b1[n] + b2[n];
  }
}

// =====================================================================
// Kernel 2: text gate  gate[b][j] = sigmoid(relu(txt@Wg1^T+b1)@Wg2^T+b2)
// =====================================================================
__global__ __launch_bounds__(256)
void k_gate(const float* __restrict__ txt,
            const float* __restrict__ Wg1, const float* __restrict__ bg1,
            const float* __restrict__ Wg2, const float* __restrict__ bg2,
            float* __restrict__ gate) {
  __shared__ float t1[HDIM];
  const int b = blockIdx.x, j = threadIdx.x;
  const float* tb = txt + (long long)b * DTXT;
  float acc = bg1[j];
  const float* w = Wg1 + (long long)j * DTXT;
  for (int k = 0; k < DTXT; k++) acc += tb[k] * w[k];
  t1[j] = fmaxf(acc, 0.f);
  __syncthreads();
  float a2 = bg2[j];
  const float* w2 = Wg2 + (long long)j * HDIM;
  for (int k = 0; k < HDIM; k++) a2 += t1[k] * w2[k];
  gate[b * HDIM + j] = sigf(a2);
}

// =====================================================================
// Kernel 3: video projection + gating (WMMA bf16, f32 accumulate)
//   xfeat[m][n] = bf16( (sum_d x[m,d]*Wvp[n,d] + bvp[n]) * gate[m/512, n] )
//   block tile 128x64, 8 waves of 32x32, K chunks of 32,
//   register double-buffered float2 staging (DIN is even -> 8B aligned).
// =====================================================================
__device__ inline void ld_stageA(const float* __restrict__ x, int m0, int k0,
                                 int tid, float2* a) {
  if (k0 + 32 <= DIN) {
#pragma unroll
    for (int e = 0; e < 8; e++) {
      int p = tid * 8 + e;                       // pair index 0..2047
      int r = p >> 4, c2 = (p & 15) * 2;
      a[e] = *(const float2*)&x[(long long)(m0 + r) * DIN + k0 + c2];
    }
  } else {                                        // tail chunk (2 valid cols)
#pragma unroll
    for (int e = 0; e < 8; e++) {
      int p = tid * 8 + e;
      int r = p >> 4, c2 = (p & 15) * 2;
      float2 v; v.x = 0.f; v.y = 0.f;
      if (k0 + c2 + 1 < DIN)
        v = *(const float2*)&x[(long long)(m0 + r) * DIN + k0 + c2];
      a[e] = v;
    }
  }
}
__device__ inline void ld_stageB(const float* __restrict__ Wvp, int n0, int k0,
                                 int tid, float2* b) {
  if (k0 + 32 <= DIN) {
#pragma unroll
    for (int e = 0; e < 4; e++) {
      int p = tid * 4 + e;                       // pair index 0..1023
      int r = p >> 4, c2 = (p & 15) * 2;
      b[e] = *(const float2*)&Wvp[(long long)(n0 + r) * DIN + k0 + c2];
    }
  } else {
#pragma unroll
    for (int e = 0; e < 4; e++) {
      int p = tid * 4 + e;
      int r = p >> 4, c2 = (p & 15) * 2;
      float2 v; v.x = 0.f; v.y = 0.f;
      if (k0 + c2 + 1 < DIN)
        v = *(const float2*)&Wvp[(long long)(n0 + r) * DIN + k0 + c2];
      b[e] = v;
    }
  }
}

__global__ __launch_bounds__(256)
void k_vproj(const float* __restrict__ x,
             const float* __restrict__ Wvp, const float* __restrict__ bvp,
             const float* __restrict__ gate,
             unsigned short* __restrict__ xfeat) {
  __shared__ __align__(16) unsigned short As[128][40];  // 128x32 bf16 (+pad)
  __shared__ __align__(16) unsigned short Bs[64][40];   // 64x32  bf16 (+pad)

  const int tid  = threadIdx.x;
  const int lane = tid & 31;
  const int wave = tid >> 5;
  const int m0 = blockIdx.x * 128;
  const int n0 = blockIdx.y * 64;
  const int batch = m0 >> 9;               // 512 rows/batch, tile never crosses
  const int wm = (wave >> 1) * 32;
  const int wn = (wave & 1) * 32;
  const int kl = (lane < 16) ? 0 : 8;

  FragC acc[2][2];
#pragma unroll
  for (int mi = 0; mi < 2; mi++)
#pragma unroll
    for (int ni = 0; ni < 2; ni++)
#pragma unroll
      for (int e = 0; e < 8; e++) acc[mi][ni].f[e] = 0.f;

  float2 aReg[8], bReg[4];
  ld_stageA(x, m0, 0, tid, aReg);
  ld_stageB(Wvp, n0, 0, tid, bReg);

  for (int ch = 0; ch < NCHUNK; ch++) {
    // commit staged registers to LDS (packed bf16 pair stores)
#pragma unroll
    for (int e = 0; e < 8; e++) {
      int p = tid * 8 + e;
      int r = p >> 4, c2 = (p & 15) * 2;
      *(unsigned int*)&As[r][c2] = pack_bf2(aReg[e].x, aReg[e].y);
    }
#pragma unroll
    for (int e = 0; e < 4; e++) {
      int p = tid * 4 + e;
      int r = p >> 4, c2 = (p & 15) * 2;
      *(unsigned int*)&Bs[r][c2] = pack_bf2(bReg[e].x, bReg[e].y);
    }
    __syncthreads();

    // prefetch next chunk while WMMA consumes this one
    if (ch + 1 < NCHUNK) {
      ld_stageA(x, m0, (ch + 1) * 32, tid, aReg);
      ld_stageB(Wvp, n0, (ch + 1) * 32, tid, bReg);
    }

    FragAB a[2], bfr[2];
#pragma unroll
    for (int mi = 0; mi < 2; mi++) {
      const unsigned short* ap = &As[wm + mi * 16 + (lane & 15)][0];
      a[mi].q[0] = *(const v4u*)(ap + kl);
      a[mi].q[1] = *(const v4u*)(ap + kl + 16);
    }
#pragma unroll
    for (int ni = 0; ni < 2; ni++) {
      const unsigned short* bp = &Bs[wn + ni * 16 + (lane & 15)][0];
      bfr[ni].q[0] = *(const v4u*)(bp + kl);
      bfr[ni].q[1] = *(const v4u*)(bp + kl + 16);
    }
#pragma unroll
    for (int mi = 0; mi < 2; mi++)
#pragma unroll
      for (int ni = 0; ni < 2; ni++)
        acc[mi][ni].v = __builtin_amdgcn_wmma_f32_16x16x32_bf16(
            false, a[mi].v, false, bfr[ni].v, (short)0, acc[mi][ni].v,
            false, false);
    __syncthreads();
  }

  // epilogue: bias + gate, store bf16
#pragma unroll
  for (int ni = 0; ni < 2; ni++) {
    int n = n0 + wn + ni * 16 + (lane & 15);
    float bias = bvp[n];
    float g = gate[batch * HDIM + n];
#pragma unroll
    for (int mi = 0; mi < 2; mi++)
#pragma unroll
      for (int v = 0; v < 8; v++) {
        int m = m0 + wm + mi * 16 + v + ((lane < 16) ? 0 : 8);
        float val = (acc[mi][ni].f[v] + bias) * g;
        xfeat[(long long)m * HDIM + n] = f2bf(val);
      }
  }
}

// =====================================================================
// Kernel 4: bidirectional LSTM recurrence + temporal max-pool
//   one 1024-thread workgroup per direction (blockIdx.x = dir)
//   per step: G = [x_t, h_{t-1}] (64x512) @ Wcomb^T (512x1024), bf16 WMMA
//   batch split into two M=32 passes; gates exchanged via LDS;
//   c and running max live in per-thread registers for all 512 steps.
// =====================================================================
__global__ __launch_bounds__(1024)
void k_lstm(const unsigned short* __restrict__ xfeat,
            const unsigned short* __restrict__ Wcomb,
            const float* __restrict__ bcomb,
            float* __restrict__ vfeat) {
  __shared__ __align__(16) unsigned short Hs[BATCH][264];   // h_{t-1} bf16
  __shared__ __align__(16) unsigned short Gs[32][1032];     // gate preacts bf16

  const int dir  = blockIdx.x;
  const int tid  = threadIdx.x;
  const int lane = tid & 31;
  const int wave = tid >> 5;       // 32 waves, each owns 32 gate columns
  const int wn   = wave * 32;
  const int kl   = (lane < 16) ? 0 : 8;
  const unsigned short* Wd = Wcomb + (long long)dir * (NG * KREC);

  float bias_r[2];
#pragma unroll
  for (int ni = 0; ni < 2; ni++)
    bias_r[ni] = bcomb[dir * NG + wn + ni * 16 + (lane & 15)];

  for (int i = tid; i < (BATCH * 264) / 2; i += 1024)
    ((unsigned int*)&Hs[0][0])[i] = 0u;

  float c[2][8], vmax[2][8];
#pragma unroll
  for (int p = 0; p < 2; p++)
#pragma unroll
    for (int ii = 0; ii < 8; ii++) { c[p][ii] = 0.f; vmax[p][ii] = -3.4e38f; }
  __syncthreads();

  for (int step = 0; step < SEQ; step++) {
    const int t = dir ? (SEQ - 1 - step) : step;
#pragma unroll
    for (int p = 0; p < 2; p++) {
      const int mp = p * 32;
      FragC acc[2][2];
#pragma unroll
      for (int mi = 0; mi < 2; mi++)
#pragma unroll
        for (int ni = 0; ni < 2; ni++)
#pragma unroll
          for (int e = 0; e < 8; e++) acc[mi][ni].f[e] = 0.f;

      for (int kt = 0; kt < 16; kt++) {
        FragAB a[2], bfr[2];
#pragma unroll
        for (int mi = 0; mi < 2; mi++) {
          int rowA = mp + mi * 16 + (lane & 15);        // batch index
          if (kt < 8) {   // x_feat part, straight from global (L2 resident)
            const unsigned short* ap =
                xfeat + ((long long)(rowA * SEQ + t)) * HDIM + kt * 32 + kl;
            a[mi].q[0] = *(const v4u*)ap;
            a[mi].q[1] = *(const v4u*)(ap + 16);
          } else {        // h part, from LDS
            const unsigned short* ap = &Hs[rowA][(kt - 8) * 32 + kl];
            a[mi].q[0] = *(const v4u*)ap;
            a[mi].q[1] = *(const v4u*)(ap + 16);
          }
        }
#pragma unroll
        for (int ni = 0; ni < 2; ni++) {
          const unsigned short* bp =
              Wd + (long long)(wn + ni * 16 + (lane & 15)) * KREC + kt * 32 + kl;
          bfr[ni].q[0] = *(const v4u*)bp;
          bfr[ni].q[1] = *(const v4u*)(bp + 16);
        }
#pragma unroll
        for (int mi = 0; mi < 2; mi++)
#pragma unroll
          for (int ni = 0; ni < 2; ni++)
            acc[mi][ni].v = __builtin_amdgcn_wmma_f32_16x16x32_bf16(
                false, a[mi].v, false, bfr[ni].v, (short)0, acc[mi][ni].v,
                false, false);
      }

      // write gate pre-activations (+bias) to LDS
#pragma unroll
      for (int ni = 0; ni < 2; ni++) {
        int n = wn + ni * 16 + (lane & 15);
#pragma unroll
        for (int mi = 0; mi < 2; mi++)
#pragma unroll
          for (int v = 0; v < 8; v++) {
            int lm = mi * 16 + v + ((lane < 16) ? 0 : 8);
            Gs[lm][n] = f2bf(acc[mi][ni].f[v] + bias_r[ni]);
          }
      }
      __syncthreads();

      // pointwise LSTM cell for batches [mp, mp+32): 8 units per thread
#pragma unroll
      for (int ii = 0; ii < 8; ii++) {
        int u = ii * 1024 + tid;          // 0..8191
        int lm = u >> 8, j = u & 255;
        float gi = bf2f(Gs[lm][j]);
        float gf = bf2f(Gs[lm][256 + j]);
        float gg = bf2f(Gs[lm][512 + j]);
        float go = bf2f(Gs[lm][768 + j]);
        float I = sigf(gi), F = sigf(gf), G = tanhf(gg), O = sigf(go);
        float cc = F * c[p][ii] + I * G;
        c[p][ii] = cc;
        float h = O * tanhf(cc);
        Hs[mp + lm][j] = f2bf(h);
        vmax[p][ii] = fmaxf(vmax[p][ii], h);
      }
      __syncthreads();
    }
  }

  // v_feat[b][dir*256 + j] = running max
#pragma unroll
  for (int p = 0; p < 2; p++)
#pragma unroll
    for (int ii = 0; ii < 8; ii++) {
      int u = ii * 1024 + tid;
      int b = p * 32 + (u >> 8), j = u & 255;
      vfeat[(long long)b * (2 * HDIM) + dir * HDIM + j] = vmax[p][ii];
    }
}

// =====================================================================
// Kernel 5: late fusion  out[b] = v@Wu^T + bu + sum(v * (txt@Wtl^T+btl))
// =====================================================================
__global__ __launch_bounds__(256)
void k_final(const float* __restrict__ vfeat, const float* __restrict__ txt,
             const float* __restrict__ Wu, const float* __restrict__ bu,
             const float* __restrict__ Wtl, const float* __restrict__ btl,
             float* __restrict__ out) {
  __shared__ float red[256];
  const int b = blockIdx.x, tidx = threadIdx.x;
  const float* tb = txt + (long long)b * DTXT;
  const float* vb = vfeat + (long long)b * (2 * HDIM);
  float acc = 0.f;
  for (int n = tidx; n < 2 * HDIM; n += 256) {
    float tl = btl[n];
    const float* w = Wtl + (long long)n * DTXT;
    for (int k = 0; k < DTXT; k++) tl += tb[k] * w[k];
    acc += vb[n] * (Wu[n] + tl);
  }
  red[tidx] = acc;
  __syncthreads();
  for (int s = 128; s > 0; s >>= 1) {
    if (tidx < s) red[tidx] += red[tidx + s];
    __syncthreads();
  }
  if (tidx == 0) out[b] = red[0] + bu[0];
}

// =====================================================================
extern "C" void kernel_launch(void* const* d_in, const int* in_sizes, int n_in,
                              void* d_out, int out_size, void* d_ws,
                              size_t ws_size, hipStream_t stream) {
  (void)in_sizes; (void)n_in; (void)out_size; (void)ws_size;
  const float* x     = (const float*)d_in[0];
  const float* txt   = (const float*)d_in[1];
  const float* W_vp  = (const float*)d_in[2];
  const float* b_vp  = (const float*)d_in[3];
  const float* W_g1  = (const float*)d_in[4];
  const float* b_g1  = (const float*)d_in[5];
  const float* W_g2  = (const float*)d_in[6];
  const float* b_g2  = (const float*)d_in[7];
  const float* Wih_f = (const float*)d_in[8];
  const float* Whh_f = (const float*)d_in[9];
  const float* bih_f = (const float*)d_in[10];
  const float* bhh_f = (const float*)d_in[11];
  const float* Wih_b = (const float*)d_in[12];
  const float* Whh_b = (const float*)d_in[13];
  const float* bih_b = (const float*)d_in[14];
  const float* bhh_b = (const float*)d_in[15];
  const float* W_u   = (const float*)d_in[16];
  const float* b_u   = (const float*)d_in[17];
  const float* W_tl  = (const float*)d_in[18];
  const float* b_tl  = (const float*)d_in[19];
  float* out = (float*)d_out;

  // workspace layout (~19.1 MB)
  char* w = (char*)d_ws;
  float*          gate  = (float*)w;          w += (size_t)BATCH * HDIM * 4;
  unsigned short* xfeat = (unsigned short*)w; w += (size_t)BS * HDIM * 2;
  unsigned short* Wcomb = (unsigned short*)w; w += (size_t)2 * NG * KREC * 2;
  float*          bcomb = (float*)w;          w += (size_t)2 * NG * 4;
  float*          vfeat = (float*)w;          w += (size_t)BATCH * 2 * HDIM * 4;

  k_prep<<<1024, 256, 0, stream>>>(Wih_f, Whh_f, bih_f, bhh_f,
                                   Wih_b, Whh_b, bih_b, bhh_b, Wcomb, bcomb);
  k_gate<<<BATCH, 256, 0, stream>>>(txt, W_g1, b_g1, W_g2, b_g2, gate);
  k_vproj<<<dim3(BS / 128, HDIM / 64), 256, 0, stream>>>(x, W_vp, b_vp, gate,
                                                         xfeat);
  k_lstm<<<2, 1024, 0, stream>>>(xfeat, Wcomb, bcomb, vfeat);
  k_final<<<BATCH, 256, 0, stream>>>(vfeat, txt, W_u, b_u, W_tl, b_tl, out);
}
